// SimpleSSMBlock_2327872274722
// MI455X (gfx1250) — matrix operand order repaired and checked by reading, not compiled, not run on hardware
//
#include <hip/hip_runtime.h>
#include <hip/hip_bf16.h>

// ---------------------------------------------------------------------------
// Problem constants (from reference): B=4, S=2048, D=1024, N=16
// ---------------------------------------------------------------------------
#define B_ 4
#define S_ 2048
#define D_ 1024
#define N_ 16
#define ROWS_ (B_ * S_)            // 8192 rows for LN / GEMM

typedef __attribute__((ext_vector_type(8)))  float  v8f;
typedef __attribute__((ext_vector_type(16))) __bf16 v16bf;

union FragBF16 { v16bf v; uint4 q[2]; };

// bf16 helpers (round-to-nearest-even), avoiding scalar __bf16 conversions
__device__ __forceinline__ unsigned short f2bf(float f) {
    unsigned int u = __builtin_bit_cast(unsigned int, f);
    unsigned int r = u + 0x7FFFu + ((u >> 16) & 1u);
    return (unsigned short)(r >> 16);
}
__device__ __forceinline__ float bf2f(unsigned short h) {
    unsigned int u = ((unsigned int)h) << 16;
    return __builtin_bit_cast(float, u);
}

// Generic shared pointer -> LDS byte offset (flat LDS aperture keeps the
// byte offset in addr[31:0] per the CDNA5 aperture scheme).
__device__ __forceinline__ unsigned lds_off(const void* p) {
    return (unsigned)(unsigned long long)p;
}

// CDNA5 async global->LDS copy, 16 bytes per lane, tracked by ASYNCcnt.
// GV addressing: VDST = LDS byte-address VGPR, VADDR = 64-bit address pair.
__device__ __forceinline__ void async_ld_b128(unsigned lds_byte_off,
                                              unsigned long long gaddr) {
    asm volatile("global_load_async_to_lds_b128 %0, %1, off"
                 :: "v"(lds_byte_off), "v"(gaddr)
                 : "memory");
}

// ---------------------------------------------------------------------------
// Kernel 1: sequential SSM scan.
// One channel (b,d) per 16-lane half-wave; 2 channels per wave32.
// lane n holds row n of A[d] (16 VGPRs), h[n], Bm[d][n], Cm[d][n].
// h_new[n] = sum_m A[n][m] * shfl(h, m)  (16 FMA + 16 shfl / step / lane)
// y_t = sum_n Cm[n] h[n] + Dm x_t       (xor-shuffle reduction)
// ---------------------------------------------------------------------------
__global__ __launch_bounds__(256) void ssm_scan_kernel(
    const float* __restrict__ x,   // [B,S,D]
    const float* __restrict__ A,   // [D,N,N]
    const float* __restrict__ Bm,  // [D,N]
    const float* __restrict__ Cm,  // [D,N]
    const float* __restrict__ Dm,  // [D]
    float* __restrict__ y_raw)     // [B,S,D]
{
    const int ch   = blockIdx.x * 16 + (threadIdx.x >> 4);  // 0 .. B*D-1
    const int lane = threadIdx.x & 15;                      // n within channel
    const int b = ch >> 10;       // ch / D_
    const int d = ch & (D_ - 1);  // ch % D_

    float Arow[N_];
    const float* Ad = A + (size_t)d * (N_ * N_) + lane * N_;
#pragma unroll
    for (int m = 0; m < N_; ++m) Arow[m] = Ad[m];

    const float bm = Bm[d * N_ + lane];
    const float cm = Cm[d * N_ + lane];
    const float dm = Dm[d];

    const float* xp = x     + (size_t)b * S_ * D_ + d;
    float*       yp = y_raw + (size_t)b * S_ * D_ + d;

    float h = 0.0f;
    for (int t = 0; t < S_; ++t) {
        // pull the x stream ahead of the serial recurrence
        if (t + 16 < S_) __builtin_prefetch(xp + (size_t)(t + 16) * D_, 0, 3);

        const float xv = xp[(size_t)t * D_];
        float hn = 0.0f;
#pragma unroll
        for (int m = 0; m < N_; ++m)
            hn = fmaf(Arow[m], __shfl(h, m, 16), hn);
        h = fmaf(bm, xv, hn);            // h = A h_prev + Bm * x_t

        float p = cm * h;                // C . h  (16-lane reduction)
#pragma unroll
        for (int off = 8; off; off >>= 1) p += __shfl_xor(p, off, 16);

        if (lane == 0) yp[(size_t)t * D_] = fmaf(dm, xv, p);
    }
}

// ---------------------------------------------------------------------------
// Kernel 2: split W_out into bf16 hi/lo pair (one-time, 4 MB -> 2x2 MB)
// ---------------------------------------------------------------------------
__global__ __launch_bounds__(256) void w_split_kernel(
    const float* __restrict__ W,
    unsigned short* __restrict__ Whi,
    unsigned short* __restrict__ Wlo)
{
    int i = blockIdx.x * 256 + threadIdx.x;
    float f = W[i];
    unsigned short hi = f2bf(f);
    unsigned short lo = f2bf(f - bf2f(hi));
    Whi[i] = hi;
    Wlo[i] = lo;
}

// ---------------------------------------------------------------------------
// Kernel 3: LayerNorm (over D) + exact GELU + bf16 hi/lo split.
// One row (1024 elems) per 256-thread block; 4 elems per thread.
// ---------------------------------------------------------------------------
__global__ __launch_bounds__(256) void ln_gelu_split_kernel(
    const float* __restrict__ y_raw,   // [ROWS, D]
    const float* __restrict__ ln_w,    // [D]
    const float* __restrict__ ln_b,    // [D]
    unsigned short* __restrict__ yn_hi,
    unsigned short* __restrict__ yn_lo)
{
    __shared__ float red[8];
    const int row = blockIdx.x;
    const int tid = threadIdx.x;
    const float* yr = y_raw + (size_t)row * D_;

    float v[4];
    float s = 0.0f;
#pragma unroll
    for (int i = 0; i < 4; ++i) { v[i] = yr[tid + 256 * i]; s += v[i]; }

#pragma unroll
    for (int off = 16; off; off >>= 1) s += __shfl_xor(s, off, 32);
    if ((tid & 31) == 0) red[tid >> 5] = s;
    __syncthreads();
    float tot = 0.0f;
#pragma unroll
    for (int i = 0; i < 8; ++i) tot += red[i];
    const float mu = tot * (1.0f / D_);
    __syncthreads();   // before reusing red[]

    float vs = 0.0f;
#pragma unroll
    for (int i = 0; i < 4; ++i) { float dlt = v[i] - mu; vs += dlt * dlt; }
#pragma unroll
    for (int off = 16; off; off >>= 1) vs += __shfl_xor(vs, off, 32);
    if ((tid & 31) == 0) red[tid >> 5] = vs;
    __syncthreads();
    float vtot = 0.0f;
#pragma unroll
    for (int i = 0; i < 8; ++i) vtot += red[i];
    const float rinv = rsqrtf(vtot * (1.0f / D_) + 1e-5f);

#pragma unroll
    for (int i = 0; i < 4; ++i) {
        const int col = tid + 256 * i;
        float u = (v[i] - mu) * rinv * ln_w[col] + ln_b[col];
        float g = 0.5f * u * (1.0f + erff(u * 0.70710678118654752f)); // exact GELU
        unsigned short hi = f2bf(g);
        unsigned short lo = f2bf(g - bf2f(hi));
        size_t o = (size_t)row * D_ + col;
        yn_hi[o] = hi;
        yn_lo[o] = lo;
    }
}

// ---------------------------------------------------------------------------
// Kernel 4: out = x + bias + gelu_ln_y @ W_out^T, via v_wmma_f32_16x16x32_bf16
// with 3xBF16 split-precision (hi*hi + hi*lo + lo*hi) for ~fp32 accuracy.
//
// Block: 256 threads (8 waves), tile M=128 x N=64, K-chunk=32.
// Waves laid out 4(M) x 2(N); each wave owns a 32x32 block = 2x2 WMMA tiles.
// LDS tiles stored [row][k], stride 40 bf16 (80B, 16B-aligned, bank-safe).
//
// Staging uses CDNA5 async global->LDS copies (ASYNCcnt) with double
// buffering: chunk c+1 streams into LDS while chunk c feeds the WMMAs.
// Every thread issues exactly 6 async b128 copies per chunk, so
// `s_wait_asynccnt 0x6` after issuing the next chunk == previous chunk done.
// ---------------------------------------------------------------------------
#define TM 128
#define TN 64
#define KC 32
#define LDSS 40
#define NCHUNK (D_ / KC)

__global__ __launch_bounds__(256) void gemm_wmma_kernel(
    const unsigned short* __restrict__ Ahg,  // yn_hi  [ROWS][D]
    const unsigned short* __restrict__ Alg,  // yn_lo  [ROWS][D]
    const unsigned short* __restrict__ Bhg,  // W_hi   [D][D]  (row j = output col)
    const unsigned short* __restrict__ Blg,  // W_lo   [D][D]
    const float* __restrict__ x,             // [ROWS][D] residual
    const float* __restrict__ b_out,         // [D]
    float* __restrict__ out)                 // [ROWS][D]
{
    __shared__ unsigned short sAh[2][TM * LDSS];
    __shared__ unsigned short sAl[2][TM * LDSS];
    __shared__ unsigned short sBh[2][TN * LDSS];
    __shared__ unsigned short sBl[2][TN * LDSS];

    const int tid = threadIdx.x;
    const int bm0 = blockIdx.x * TM;
    const int bn0 = blockIdx.y * TN;

    const int w    = tid >> 5;          // wave id 0..7
    const int L    = tid & 31;          // lane
    const int wm   = (w & 3) * 32;      // wave M offset in tile
    const int wn   = (w >> 2) * 32;     // wave N offset in tile
    const int lm   = L & 15;            // row (A) / col (B) within 16
    const int half = L >> 4;            // lane-half selector

    // issue one K-chunk's tile loads (6 async b128 per thread, uniform)
    auto issue_chunk = [&](int kc, int buf) {
#pragma unroll
        for (int ii = 0; ii < 2; ++ii) {           // A tile: 512 uint4 = 2/thread
            const int i = tid + ii * 256;
            const int r = i >> 2, q = i & 3;
            const size_t ge = (size_t)(bm0 + r) * D_ + kc + q * 8;
            const int    le = r * LDSS + q * 8;
            async_ld_b128(lds_off(&sAh[buf][le]),
                          (unsigned long long)(Ahg + ge));
            async_ld_b128(lds_off(&sAl[buf][le]),
                          (unsigned long long)(Alg + ge));
        }
        {                                           // B tile: 256 uint4 = 1/thread
            const int r = tid >> 2, q = tid & 3;
            const size_t ge = (size_t)(bn0 + r) * D_ + kc + q * 8;
            const int    le = r * LDSS + q * 8;
            async_ld_b128(lds_off(&sBh[buf][le]),
                          (unsigned long long)(Bhg + ge));
            async_ld_b128(lds_off(&sBl[buf][le]),
                          (unsigned long long)(Blg + ge));
        }
    };

    v8f acc[2][2] = {};

    issue_chunk(0, 0);   // prologue: chunk 0 in flight

    for (int c = 0; c < NCHUNK; ++c) {
        const int buf = c & 1;
        if (c + 1 < NCHUNK) {
            issue_chunk((c + 1) * KC, (c + 1) & 1);   // overlap next chunk
            asm volatile("s_wait_asynccnt 0x6" ::: "memory");  // chunk c landed
        } else {
            asm volatile("s_wait_asynccnt 0x0" ::: "memory");
        }
        __syncthreads();   // all waves' async fills of buf are visible

        const unsigned short* pAh = sAh[buf];
        const unsigned short* pAl = sAl[buf];
        const unsigned short* pBh = sBh[buf];
        const unsigned short* pBl = sBl[buf];

#pragma unroll
        for (int mi = 0; mi < 2; ++mi) {
            // A fragment, 16x32 bf16. ISA layout: lane (m=L%16, half=L/16),
            // element e <-> k = (e<8 ? 0 : 16) + half*8 + (e%8)
            // => two contiguous 16B chunks at k = half*8 and k = 16 + half*8.
            const int ar = (wm + mi * 16 + lm) * LDSS;
            FragBF16 ah, al;
            ah.q[0] = *(const uint4*)(&pAh[ar + half * 8]);
            ah.q[1] = *(const uint4*)(&pAh[ar + 16 + half * 8]);
            al.q[0] = *(const uint4*)(&pAl[ar + half * 8]);
            al.q[1] = *(const uint4*)(&pAl[ar + 16 + half * 8]);

#pragma unroll
            for (int ni = 0; ni < 2; ++ni) {
                // B fragment, 32x16 bf16. lane col n=L%16; elem e <-> k=(e<8 ? 0 : 16)+half*8+(e%8).
                const int br = (wn + ni * 16 + lm) * LDSS;
                FragBF16 bh, bl;
                bh.q[0] = *(const uint4*)(&pBh[br + half * 8]);
                bh.q[1] = *(const uint4*)(&pBh[br + 16 + half * 8]);
                bl.q[0] = *(const uint4*)(&pBl[br + half * 8]);
                bl.q[1] = *(const uint4*)(&pBl[br + 16 + half * 8]);

                v8f cacc = acc[mi][ni];
                cacc = __builtin_amdgcn_wmma_f32_16x16x32_bf16(
                           false, ah.v, false, bh.v, (short)0, cacc, false, false);
                cacc = __builtin_amdgcn_wmma_f32_16x16x32_bf16(
                           false, ah.v, false, bl.v, (short)0, cacc, false, false);
                cacc = __builtin_amdgcn_wmma_f32_16x16x32_bf16(
                           false, al.v, false, bh.v, (short)0, cacc, false, false);
                acc[mi][ni] = cacc;
            }
        }
        __syncthreads();   // buf may be overwritten by chunk c+2's async fill
    }

    // ---- epilogue: out = acc + bias + residual ----
    // C/D layout: lane L -> col n = L%16; VGPR v -> row m = v + (L/16)*8
#pragma unroll
    for (int mi = 0; mi < 2; ++mi) {
#pragma unroll
        for (int ni = 0; ni < 2; ++ni) {
            const int gj = bn0 + wn + ni * 16 + lm;
            const float bb = b_out[gj];
#pragma unroll
            for (int v = 0; v < 8; ++v) {
                const int gi = bm0 + wm + mi * 16 + v + half * 8;
                const size_t o = (size_t)gi * D_ + gj;
                out[o] = acc[mi][ni][v] + bb + x[o];
            }
        }
    }
}

// ---------------------------------------------------------------------------
// Host-side launcher
// ---------------------------------------------------------------------------
extern "C" void kernel_launch(void* const* d_in, const int* in_sizes, int n_in,
                              void* d_out, int out_size, void* d_ws, size_t ws_size,
                              hipStream_t stream) {
    const float* x     = (const float*)d_in[0];
    const float* A     = (const float*)d_in[1];
    const float* Bm    = (const float*)d_in[2];
    const float* Cm    = (const float*)d_in[3];
    const float* Dm    = (const float*)d_in[4];
    const float* ln_w  = (const float*)d_in[5];
    const float* ln_b  = (const float*)d_in[6];
    const float* W_out = (const float*)d_in[7];
    const float* b_out = (const float*)d_in[8];
    float* out = (float*)d_out;

    // workspace layout (bytes)
    char* ws = (char*)d_ws;
    float*          y_raw = (float*)ws;                                   // 33.5 MB
    unsigned short* yn_hi = (unsigned short*)(ws + (size_t)ROWS_ * D_ * 4);             // 16.8 MB
    unsigned short* yn_lo = (unsigned short*)(ws + (size_t)ROWS_ * D_ * 4
                                                 + (size_t)ROWS_ * D_ * 2);             // 16.8 MB
    unsigned short* w_hi  = (unsigned short*)(ws + (size_t)ROWS_ * D_ * 4
                                                 + (size_t)ROWS_ * D_ * 4);             // 2 MB
    unsigned short* w_lo  = (unsigned short*)(ws + (size_t)ROWS_ * D_ * 4
                                                 + (size_t)ROWS_ * D_ * 4
                                                 + (size_t)D_ * D_ * 2);                // 2 MB

    // 1) split W_out into bf16 hi/lo (independent of scan)
    w_split_kernel<<<(D_ * D_) / 256, 256, 0, stream>>>(W_out, w_hi, w_lo);

    // 2) sequential SSM scan: 4096 channels, 16 per block
    ssm_scan_kernel<<<(B_ * D_) / 16, 256, 0, stream>>>(x, A, Bm, Cm, Dm, y_raw);

    // 3) layernorm + exact gelu + bf16 split, one row per block
    ln_gelu_split_kernel<<<ROWS_, 256, 0, stream>>>(y_raw, ln_w, ln_b, yn_hi, yn_lo);

    // 4) WMMA GEMM + bias + residual (async double-buffered staging)
    dim3 grid(ROWS_ / TM, D_ / TN);
    gemm_wmma_kernel<<<grid, 256, 0, stream>>>(yn_hi, yn_lo, w_hi, w_lo, x, b_out, out);
}
